// FeatExtractor_70729521430971
// MI455X (gfx1250) — matrix operand order repaired and checked by reading, not compiled, not run on hardware
//
#include <hip/hip_runtime.h>
#include <hip/hip_bf16.h>

// ---------------------------------------------------------------------------
// CDNA5 (gfx1250) fused MLP + BatchNorm + gate + segment-sum
//   pass1: GEMM1 (bf16 WMMA) -> column sum/sumsq (stats for BN1)
//   fin1 : fold mean/var/gamma/beta -> per-col affine a1,s1
//   pass2: GEMM1 -> BN1 -> SiLU -> GEMM2 -> stats for BN2
//   fin2 : a2,s2
//   pass3: GEMM1 -> BN1 -> SiLU -> GEMM2 -> BN2 -> sigmoid*softplus ->
//          global_atomic_add_f32 scatter into out[vid]
// b1/b2 provably cancel inside the BatchNorms and are dropped.
// ---------------------------------------------------------------------------

typedef __bf16 bf16;
typedef bf16  v16bf __attribute__((ext_vector_type(16)));
typedef bf16  v8bf  __attribute__((ext_vector_type(8)));
typedef float v8f   __attribute__((ext_vector_type(8)));

#define E_TOT   1000000
#define CHEM    47
#define H1      128
#define H2      256
#define NSEG    62500
#define KP      64          // K of GEMM1 padded 47 -> 64 (2 x k32 steps)
#define NCHUNK  62500       // E / 16 row-chunks (exact)
#define EPS_BN  1e-5f

// ---- WMMA helpers ---------------------------------------------------------

__device__ __forceinline__ v8f wmma_bf16(v16bf a, v16bf b, v8f c) {
  // v_wmma_f32_16x16x32_bf16, f32 accumulate
  return __builtin_amdgcn_wmma_f32_16x16x32_bf16(false, a, false, b,
                                                 (short)0, c, false, false);
}

// A-fragment (16x32 bf16, MxK): lanes 0-15 row M=lane hold K [kb,kb+8)+[kb+16,kb+24)
// with kb=0; lanes 16-31 same rows with kb=8.  base = row-major [16][ldk] in LDS.
__device__ __forceinline__ v16bf load_Afrag(const bf16* base, int ldk, int lane, int kofs) {
  int m  = lane & 15;
  int kb = kofs + ((lane >> 4) << 3);
  v8bf lo = *(const v8bf*)(base + m * ldk + kb);
  v8bf hi = *(const v8bf*)(base + m * ldk + kb + 16);
  v16bf r;
#pragma unroll
  for (int i = 0; i < 8; ++i) { r[i] = lo[i]; r[i + 8] = hi[i]; }
  return r;
}

// B-fragment (32x16 bf16, KxN): column n = col0 + (lane&15); lanes 0-15 hold K
// [kb,kb+8)+[kb+16,kb+24) with kb=kofs; lanes 16-31 kb=kofs+8.  w = K-contig [N][ldk].
__device__ __forceinline__ v16bf load_Bfrag(const bf16* w, int ldk, int col0,
                                            int lane, int kofs) {
  int n  = col0 + (lane & 15);
  int kb = kofs + ((lane >> 4) << 3);
  v8bf lo = *(const v8bf*)(w + n * ldk + kb);
  v8bf hi = *(const v8bf*)(w + n * ldk + kb + 16);
  v16bf r;
#pragma unroll
  for (int i = 0; i < 8; ++i) { r[i] = lo[i]; r[i + 8] = hi[i]; }
  return r;
}

// Stage 16 chem rows (47 f32, zero-padded to 64) as bf16 into wave-private LDS.
__device__ __forceinline__ void stage_chem(const float* __restrict__ chem,
                                           bf16* A, long e0, int lane) {
#pragma unroll 4
  for (int i = lane; i < 16 * KP; i += 32) {
    int r = i >> 6, k = i & 63;
    float v = (k < CHEM) ? chem[(e0 + r) * CHEM + k] : 0.0f;
    A[i] = (bf16)v;
  }
}

__device__ __forceinline__ v8f gemm1_tile(const bf16* A, const bf16* sW1,
                                          int t, int lane) {
  v8f acc = {};
#pragma unroll
  for (int ks = 0; ks < 2; ++ks)
    acc = wmma_bf16(load_Afrag(A, KP, lane, ks * 32),
                    load_Bfrag(sW1, KP, t * 16, lane, ks * 32), acc);
  return acc;
}

// ---- prep: zero output/stats, build bf16 transposed-padded weights --------

__global__ __launch_bounds__(256) void kprep(float* __restrict__ out,
                                             float* __restrict__ stats,
                                             bf16* __restrict__ W1t,
                                             const float* __restrict__ W1,
                                             bf16* __restrict__ W2t,
                                             const float* __restrict__ W2) {
  long i0 = (long)blockIdx.x * 256 + threadIdx.x;
  long gs = (long)gridDim.x * 256;
  for (long i = i0; i < (long)NSEG * H1; i += gs) out[i] = 0.0f;
  for (long i = i0; i < 768; i += gs) stats[i] = 0.0f;
  for (long i = i0; i < H1 * KP; i += gs) {            // W1t[c][k], k padded
    int c = (int)(i >> 6), k = (int)(i & 63);
    W1t[i] = (bf16)((k < CHEM) ? W1[(long)k * H1 + c] : 0.0f);
  }
  for (long i = i0; i < (long)H2 * H1; i += gs) {      // W2t[c][k]
    int c = (int)(i >> 7), k = (int)(i & 127);
    W2t[i] = (bf16)W2[(long)k * H2 + c];
  }
}

// ---- pass1: GEMM1 + column stats ------------------------------------------

__global__ __launch_bounds__(256) void k1_stats1(const float* __restrict__ chem,
                                                 const bf16* __restrict__ W1t,
                                                 float* __restrict__ sum1,
                                                 float* __restrict__ sq1) {
  __shared__ __align__(32) bf16 sW1[H1 * KP];
  __shared__ __align__(32) bf16 sA[8][16 * KP];
  int tid = threadIdx.x, lane = tid & 31, wave = tid >> 5;
  { const uint4* s = (const uint4*)W1t; uint4* d = (uint4*)sW1;
    for (int i = tid; i < H1 * KP * 2 / 16; i += 256) d[i] = s[i]; }
  __syncthreads();

  float lsum[8], lsq[8];
#pragma unroll
  for (int i = 0; i < 8; ++i) { lsum[i] = 0.0f; lsq[i] = 0.0f; }

  int stride = gridDim.x * 8;
  for (int chunk = blockIdx.x * 8 + wave; chunk < NCHUNK; chunk += stride) {
    long e0 = (long)chunk * 16;
    bf16* A = sA[wave];
    stage_chem(chem, A, e0, lane);
    asm volatile("s_wait_dscnt 0" ::: "memory");
    long en = e0 + (long)stride * 16;                 // prefetch next chunk
    if (en < E_TOT) __builtin_prefetch(&chem[en * CHEM + lane * 24], 0, 1);
#pragma unroll
    for (int t = 0; t < 8; ++t) {
      v8f acc = gemm1_tile(A, sW1, t, lane);
#pragma unroll
      for (int v = 0; v < 8; ++v) { float x = acc[v]; lsum[t] += x; lsq[t] += x * x; }
    }
  }
#pragma unroll
  for (int t = 0; t < 8; ++t) {
    float s = lsum[t] + __shfl_down(lsum[t], 16, 32);
    float q = lsq[t]  + __shfl_down(lsq[t], 16, 32);
    if (lane < 16) { atomicAdd(&sum1[t * 16 + lane], s);
                     atomicAdd(&sq1[t * 16 + lane],  q); }
  }
}

// ---- finalize BN stats into per-column affine -----------------------------

__global__ void kfin(const float* __restrict__ sum, const float* __restrict__ sq,
                     const float* __restrict__ gamma, const float* __restrict__ beta,
                     float* __restrict__ a, float* __restrict__ s, int C) {
  int c = blockIdx.x * blockDim.x + threadIdx.x;
  if (c < C) {
    const float invE = 1.0f / (float)E_TOT;
    float m = sum[c] * invE;
    float v = sq[c] * invE - m * m;
    float r = rsqrtf(fmaxf(v, 0.0f) + EPS_BN);
    float ac = gamma[c] * r;
    a[c] = ac;
    s[c] = beta[c] - m * ac;
  }
}

// ---- pass2: GEMM1 -> BN1 -> SiLU -> GEMM2 -> stats2 -----------------------

__global__ __launch_bounds__(256) void k2_stats2(const float* __restrict__ chem,
                                                 const bf16* __restrict__ W1t,
                                                 const bf16* __restrict__ W2t,
                                                 const float* __restrict__ a1,
                                                 const float* __restrict__ s1,
                                                 float* __restrict__ sum2,
                                                 float* __restrict__ sq2) {
  __shared__ __align__(32) bf16 sW1[H1 * KP];
  __shared__ __align__(32) bf16 sW2[H2 * H1];
  __shared__ __align__(32) bf16 sA[8][16 * KP];
  __shared__ __align__(32) bf16 sH[8][16 * H1];
  __shared__ float sAS[2 * H1];
  int tid = threadIdx.x, lane = tid & 31, wave = tid >> 5;
  { const uint4* s = (const uint4*)W1t; uint4* d = (uint4*)sW1;
    for (int i = tid; i < H1 * KP * 2 / 16; i += 256) d[i] = s[i]; }
  { const uint4* s = (const uint4*)W2t; uint4* d = (uint4*)sW2;
    for (int i = tid; i < H2 * H1 * 2 / 16; i += 256) d[i] = s[i]; }
  for (int i = tid; i < H1; i += 256) { sAS[i] = a1[i]; sAS[H1 + i] = s1[i]; }
  __syncthreads();

  float lsum[16], lsq[16];
#pragma unroll
  for (int i = 0; i < 16; ++i) { lsum[i] = 0.0f; lsq[i] = 0.0f; }

  int stride = gridDim.x * 8;
  for (int chunk = blockIdx.x * 8 + wave; chunk < NCHUNK; chunk += stride) {
    long e0 = (long)chunk * 16;
    bf16* A  = sA[wave];
    bf16* Hs = sH[wave];
    stage_chem(chem, A, e0, lane);
    asm volatile("s_wait_dscnt 0" ::: "memory");
    long en = e0 + (long)stride * 16;
    if (en < E_TOT) __builtin_prefetch(&chem[en * CHEM + lane * 24], 0, 1);
    int m_hi = (lane >> 4) << 3;
#pragma unroll
    for (int t = 0; t < 8; ++t) {
      v8f acc = gemm1_tile(A, sW1, t, lane);
      int col = t * 16 + (lane & 15);
      float ac = sAS[col], sc = sAS[H1 + col];
#pragma unroll
      for (int v = 0; v < 8; ++v) {
        float y = fmaf(ac, acc[v], sc);
        y = y * (1.0f / (1.0f + __expf(-y)));          // SiLU
        Hs[(v + m_hi) * H1 + col] = (bf16)y;
      }
    }
    asm volatile("s_wait_dscnt 0" ::: "memory");
    v16bf a2f[4];
#pragma unroll
    for (int ks = 0; ks < 4; ++ks) a2f[ks] = load_Afrag(Hs, H1, lane, ks * 32);
#pragma unroll
    for (int t = 0; t < 16; ++t) {
      v8f acc = {};
#pragma unroll
      for (int ks = 0; ks < 4; ++ks)
        acc = wmma_bf16(a2f[ks], load_Bfrag(sW2, H1, t * 16, lane, ks * 32), acc);
#pragma unroll
      for (int v = 0; v < 8; ++v) { float x = acc[v]; lsum[t] += x; lsq[t] += x * x; }
    }
  }
#pragma unroll
  for (int t = 0; t < 16; ++t) {
    float s = lsum[t] + __shfl_down(lsum[t], 16, 32);
    float q = lsq[t]  + __shfl_down(lsq[t], 16, 32);
    if (lane < 16) { atomicAdd(&sum2[t * 16 + lane], s);
                     atomicAdd(&sq2[t * 16 + lane],  q); }
  }
}

// ---- pass3: full chain -> gate -> atomic scatter --------------------------

__global__ __launch_bounds__(256) void k3_scatter(const float* __restrict__ chem,
                                                  const bf16* __restrict__ W1t,
                                                  const bf16* __restrict__ W2t,
                                                  const float* __restrict__ a1,
                                                  const float* __restrict__ s1,
                                                  const float* __restrict__ a2,
                                                  const float* __restrict__ s2,
                                                  const int* __restrict__ vid,
                                                  float* __restrict__ out) {
  __shared__ __align__(32) bf16 sW1[H1 * KP];
  __shared__ __align__(32) bf16 sW2[H2 * H1];
  __shared__ __align__(32) bf16 sA[8][16 * KP];
  __shared__ __align__(32) bf16 sH[8][16 * H1];
  __shared__ float sAS[2 * H1 + 2 * H2];   // a1|s1|a2|s2
  int tid = threadIdx.x, lane = tid & 31, wave = tid >> 5;
  { const uint4* s = (const uint4*)W1t; uint4* d = (uint4*)sW1;
    for (int i = tid; i < H1 * KP * 2 / 16; i += 256) d[i] = s[i]; }
  { const uint4* s = (const uint4*)W2t; uint4* d = (uint4*)sW2;
    for (int i = tid; i < H2 * H1 * 2 / 16; i += 256) d[i] = s[i]; }
  for (int i = tid; i < H1; i += 256) { sAS[i] = a1[i]; sAS[H1 + i] = s1[i]; }
  for (int i = tid; i < H2; i += 256) { sAS[256 + i] = a2[i]; sAS[512 + i] = s2[i]; }
  __syncthreads();

  int stride = gridDim.x * 8;
  for (int chunk = blockIdx.x * 8 + wave; chunk < NCHUNK; chunk += stride) {
    long e0 = (long)chunk * 16;
    bf16* A  = sA[wave];
    bf16* Hs = sH[wave];
    stage_chem(chem, A, e0, lane);
    asm volatile("s_wait_dscnt 0" ::: "memory");
    long en = e0 + (long)stride * 16;
    if (en < E_TOT) __builtin_prefetch(&chem[en * CHEM + lane * 24], 0, 1);
    int m_hi = (lane >> 4) << 3;
#pragma unroll
    for (int t = 0; t < 8; ++t) {
      v8f acc = gemm1_tile(A, sW1, t, lane);
      int col = t * 16 + (lane & 15);
      float ac = sAS[col], sc = sAS[H1 + col];
#pragma unroll
      for (int v = 0; v < 8; ++v) {
        float y = fmaf(ac, acc[v], sc);
        y = y * (1.0f / (1.0f + __expf(-y)));
        Hs[(v + m_hi) * H1 + col] = (bf16)y;
      }
    }
    asm volatile("s_wait_dscnt 0" ::: "memory");
    v16bf a2f[4];
#pragma unroll
    for (int ks = 0; ks < 4; ++ks) a2f[ks] = load_Afrag(Hs, H1, lane, ks * 32);
    int vr[8];
#pragma unroll
    for (int v = 0; v < 8; ++v) vr[v] = vid[e0 + v + m_hi];
#pragma unroll
    for (int tp = 0; tp < 8; ++tp) {
      v8f aF = {}, aC = {};
#pragma unroll
      for (int ks = 0; ks < 4; ++ks) {
        aF = wmma_bf16(a2f[ks], load_Bfrag(sW2, H1,  tp      * 16, lane, ks * 32), aF);
        aC = wmma_bf16(a2f[ks], load_Bfrag(sW2, H1, (tp + 8) * 16, lane, ks * 32), aC);
      }
      int colF = tp * 16 + (lane & 15);
      int colC = colF + H1;
      float afc = sAS[256 + colF], sfc = sAS[512 + colF];
      float acc2 = sAS[256 + colC], scc = sAS[512 + colC];
#pragma unroll
      for (int v = 0; v < 8; ++v) {
        float f = fmaf(afc,  aF[v], sfc);
        float c = fmaf(acc2, aC[v], scc);
        float g = (1.0f / (1.0f + __expf(-f))) *
                  (fmaxf(c, 0.0f) + log1pf(__expf(-fabsf(c))));   // sigmoid*softplus
        atomicAdd(&out[(long)vr[v] * H1 + colF], g);
      }
    }
  }
}

// ---------------------------------------------------------------------------

extern "C" void kernel_launch(void* const* d_in, const int* in_sizes, int n_in,
                              void* d_out, int out_size, void* d_ws, size_t ws_size,
                              hipStream_t stream) {
  const float* chem = (const float*)d_in[0];
  const int*   vid  = (const int*)d_in[1];
  const float* W1   = (const float*)d_in[2];
  // d_in[3] = b1 : cancelled by BN1
  const float* g1   = (const float*)d_in[4];
  const float* bt1  = (const float*)d_in[5];
  const float* W2   = (const float*)d_in[6];
  // d_in[7] = b2 : cancelled by BN2
  const float* g2   = (const float*)d_in[8];
  const float* bt2  = (const float*)d_in[9];
  float* out = (float*)d_out;

  float* wsf  = (float*)d_ws;
  float* sum1 = wsf;            // 128
  float* sq1  = wsf + 128;      // 128
  float* sum2 = wsf + 256;      // 256
  float* sq2  = wsf + 512;      // 256
  float* a1   = wsf + 768;      // 128
  float* s1   = wsf + 896;      // 128
  float* a2   = wsf + 1024;     // 256
  float* s2   = wsf + 1280;     // 256
  bf16*  W1t  = (bf16*)(wsf + 1536);   // 128*64 bf16 (16 KB)
  bf16*  W2t  = W1t + H1 * KP;         // 256*128 bf16 (64 KB)

  kprep<<<2048, 256, 0, stream>>>(out, wsf, W1t, W1, W2t, W2);
  k1_stats1<<<512, 256, 0, stream>>>(chem, W1t, sum1, sq1);
  kfin<<<1, 256, 0, stream>>>(sum1, sq1, g1, bt1, a1, s1, H1);
  k2_stats2<<<512, 256, 0, stream>>>(chem, W1t, W2t, a1, s1, sum2, sq2);
  kfin<<<1, 256, 0, stream>>>(sum2, sq2, g2, bt2, a2, s2, H2);
  k3_scatter<<<512, 256, 0, stream>>>(chem, W1t, W2t, a1, s1, a2, s2, vid, out);
}
